// LTAE2d_16509854286507
// MI455X (gfx1250) — compile-verified, matrix operand order checked by reading, not run on hardware
//
#include <hip/hip_runtime.h>
#include <hip/hip_bf16.h>
#include <cstddef>
#include <cstdint>

typedef __attribute__((ext_vector_type(16))) _Float16 v16h;
typedef __attribute__((ext_vector_type(8)))  float    v8f;

#define B_ 4
#define T_ 24
#define C_ 128
#define H_ 48
#define W_ 48
#define HW_ 2304            // H_*W_
#define NH_ 16
#define DK_ 4
#define DM_ 256
#define DO_ 128
#define NPIX 9216           // B_*HW_
#define EPS_ 1e-5f

// K index inside a 16-bit WMMA A/B fragment for element e (0..15) of lane L:
// K = (e&7) + 16*(e>=8) + 8*(L>=16)   (ISA 7.12.2, 16-bit 16x32 A layout; B mirrored)
__device__ __host__ inline int frag_k(int lane, int e) {
    return (e & 7) + ((e >> 3) << 4) + ((lane >> 4) << 3);
}

// ---------------------------------------------------------------------------
// prep: qs table, vbias (PE + b_inconv), loff (PE/bias folded into logits),
// and WMMA-fragment-swizzled f16 weights.
// ---------------------------------------------------------------------------
__global__ void prep_kernel(const int* __restrict__ pos,
                            const float* __restrict__ b_inconv,
                            const float* __restrict__ Q,
                            const float* __restrict__ W_inconv,
                            const float* __restrict__ W_k,
                            const float* __restrict__ b_k,
                            const float* __restrict__ W_mlp,
                            float* __restrict__ gQS,
                            float* __restrict__ gVbias,
                            float* __restrict__ gLoff,
                            _Float16* __restrict__ gW1F,
                            _Float16* __restrict__ gW2F,
                            _Float16* __restrict__ gW3F) {
    int idx = blockIdx.x * 256 + threadIdx.x;
    if (idx < 64) {                       // qs = Q / sqrt(DK)
        gQS[idx] = Q[idx] * 0.5f;
        return;
    }
    idx -= 64;
    if (idx < B_ * T_ * DM_) {            // vbias[b,t,dm] = b_inconv + PE
        int bt = idx >> 8, dm = idx & 255, i = dm & 15;
        float denom = __powf(1000.f, (float)(i >> 1) * 0.125f);
        float ang = (float)pos[bt] / denom;
        float pe = (i & 1) ? cosf(ang) : sinf(ang);
        gVbias[idx] = b_inconv[dm] + pe;
        return;
    }
    idx -= B_ * T_ * DM_;
    if (idx < 4 * 16 * 32 * 16) {         // W1F: K=128 (4 ktiles), N=256 (16 ntiles)
        int e = idx & 15, lane = (idx >> 4) & 31, j = (idx >> 9) & 15, kk = idx >> 13;
        int K = kk * 32 + frag_k(lane, e);
        int nc = j * 16 + (lane & 15);
        gW1F[idx] = (_Float16)W_inconv[nc * C_ + K];
        return;
    }
    idx -= 4 * 16 * 32 * 16;
    if (idx < 8 * 4 * 32 * 16) {          // W2F: K=256 (8 ktiles), N=64 (4 ntiles)
        int e = idx & 15, lane = (idx >> 4) & 31, j = (idx >> 9) & 3, kk = idx >> 11;
        int K = kk * 32 + frag_k(lane, e);
        int nc = j * 16 + (lane & 15);
        gW2F[idx] = (_Float16)W_k[nc * DM_ + K];
        return;
    }
    idx -= 8 * 4 * 32 * 16;
    if (idx < 8 * 8 * 32 * 16) {          // W3F: K=256 (8 ktiles), N=128 (8 ntiles)
        int e = idx & 15, lane = (idx >> 4) & 31, j = (idx >> 9) & 7, kk = idx >> 12;
        int K = kk * 32 + frag_k(lane, e);
        int nc = j * 16 + (lane & 15);
        gW3F[idx] = (_Float16)W_mlp[nc * DM_ + K];
        return;
    }
    idx -= 8 * 8 * 32 * 16;
    if (idx < B_ * T_ * NH_) {            // loff[b,t,h] = sum_d qs*(PE-part of k + b_k)
        int h = idx & 15, bt = idx >> 4;
        float posf = (float)pos[bt];
        float acc = 0.f;
        for (int d = 0; d < 4; ++d) {
            int col = 4 * h + d;
            float kpe = 0.f;
            for (int dm = 0; dm < DM_; ++dm) {
                int i = dm & 15;
                float denom = __powf(1000.f, (float)(i >> 1) * 0.125f);
                float ang = posf / denom;
                float pe = (i & 1) ? cosf(ang) : sinf(ang);
                kpe += (b_inconv[dm] + pe) * W_k[col * DM_ + dm];
            }
            acc += (Q[col] * 0.5f) * (kpe + b_k[col]);
        }
        gLoff[idx] = acc;
        return;
    }
}

// ---------------------------------------------------------------------------
// stats: per-(pixel, group) GroupNorm mean / rstd over 8 channels x 24 steps
// ---------------------------------------------------------------------------
__global__ void stats_kernel(const float* __restrict__ x,
                             float* __restrict__ gMean,
                             float* __restrict__ gRstd) {
    int n = blockIdx.x * 256 + threadIdx.x;     // grid.x = NPIX/256
    int g = blockIdx.y;                         // 16 groups
    int b = n / HW_, hw = n - b * HW_;
    float s = 0.f, s2 = 0.f;
    for (int t = 0; t < T_; ++t)
        for (int cc = 0; cc < 8; ++cc) {
            float v = x[((size_t)(b * T_ + t) * C_ + g * 8 + cc) * HW_ + hw];
            s += v; s2 += v * v;
        }
    float m = s * (1.f / 192.f);
    float var = s2 * (1.f / 192.f) - m * m;
    gMean[n * NH_ + g] = m;
    gRstd[n * NH_ + g] = rsqrtf(var + EPS_);
}

// ---------------------------------------------------------------------------
// main: one wave per 16-pixel tile. x strips staged LDS-side with
// global_load_async_to_lds_b128 (double-issued one t ahead); all matmuls on
// v_wmma_f32_16x16x32_f16; GEMM1 strip stored in LDS in A-fragment layout so
// the keys GEMM loads each A fragment as one contiguous 32B per-lane read.
// ---------------------------------------------------------------------------
__global__ __launch_bounds__(32) void main_kernel(
        const float* __restrict__ x,
        const float* __restrict__ gGam, const float* __restrict__ gBet,
        const float* __restrict__ b_mlp,
        const float* __restrict__ gMean, const float* __restrict__ gRstd,
        const float* __restrict__ gVbias, const float* __restrict__ gLoff,
        const float* __restrict__ gQS,
        const _Float16* __restrict__ gW1F,
        const _Float16* __restrict__ gW2F,
        const _Float16* __restrict__ gW3F,
        float* __restrict__ gY) {
    __shared__ float    sMean[16][16];
    __shared__ float    sRstd[16][16];
    __shared__ float    sGam[C_], sBet[C_];
    __shared__ __align__(32) _Float16 sOutFrag[8][32][16]; // GEMM1 strip, A-frag layout
    __shared__ float    sAttn[T_][16][NH_];                // logits -> attention weights
    __shared__ float    sOacc[16][DM_];                    // weighted value accumulator
    __shared__ __align__(16) float sXs[C_][16];            // async-staged x strip [c][p]

    const int lane = threadIdx.x;
    const int n0 = blockIdx.x * 16;
    const int b = n0 / HW_;
    const int hw0 = n0 - b * HW_;
    const int p   = lane & 15;               // A-fragment row / D-tile column lane
    const int hi8 = (lane >> 4) << 3;        // half-wave offset (M rows 8..15, K +8)

    // ---- per-lane async staging addresses (8 KB strip = 16 x 512B copies) ----
    const int cRow = lane >> 2;              // 8 channel rows per instruction
    const int pxB  = (lane & 3) * 16;        // 16B chunk within 64B pixel row
    const unsigned ldsX0 = (unsigned)(uintptr_t)(&sXs[0][0]) +
                           (unsigned)cRow * 64u + (unsigned)pxB;
    const unsigned long long gX0 = (unsigned long long)(uintptr_t)x +
                           (unsigned long long)cRow * (HW_ * 4) +
                           (unsigned long long)hw0 * 4 + (unsigned long long)pxB;

    #define STAGE_STRIP(tt)                                                          \
        do {                                                                         \
            unsigned lo = ldsX0;                                                     \
            unsigned long long ga = gX0 +                                            \
                (unsigned long long)((size_t)(b * T_ + (tt)) * C_) * (HW_ * 4);      \
            for (int i_ = 0; i_ < 16; ++i_) {                                        \
                asm volatile("global_load_async_to_lds_b128 %0, %1, off"             \
                             :: "v"(lo), "v"(ga) : "memory");                        \
                lo += 512u;                                                          \
                ga += (unsigned long long)8 * HW_ * 4;                               \
            }                                                                        \
        } while (0)

    STAGE_STRIP(0);                          // kick off t=0 while we set up LDS

    for (int i = lane; i < 256; i += 32) {
        sMean[i >> 4][i & 15] = gMean[(n0 + (i >> 4)) * NH_ + (i & 15)];
        sRstd[i >> 4][i & 15] = gRstd[(n0 + (i >> 4)) * NH_ + (i & 15)];
    }
    for (int i = lane; i < C_; i += 32) { sGam[i] = gGam[i]; sBet[i] = gBet[i]; }
    for (int i = lane; i < 16 * DM_; i += 32) (&sOacc[0][0])[i] = 0.f;
    __syncthreads();

    // build normalized A fragments (GEMM1) for the staged strip
    #define BUILD_AFRAG(afrag)                                                       \
        do {                                                                         \
            for (int kk = 0; kk < 4; ++kk) {                                         \
                v16h f;                                                              \
                for (int e = 0; e < 16; ++e) {                                       \
                    int c = kk * 32 + frag_k(lane, e);                               \
                    int g = c >> 3;                                                  \
                    float xv = sXs[c][p];                                            \
                    f[e] = (_Float16)((xv - sMean[p][g]) * sRstd[p][g] * sGam[c]     \
                                      + sBet[c]);                                    \
                }                                                                    \
                afrag[kk] = f;                                                       \
            }                                                                        \
        } while (0)

    // ----- pass 1: logits -----
    for (int t = 0; t < T_; ++t) {
        asm volatile("s_wait_asynccnt 0x0" ::: "memory");   // strip t landed in LDS
        __syncthreads();

        v16h afrag[4];
        BUILD_AFRAG(afrag);
        asm volatile("s_wait_dscnt 0x0" ::: "memory");      // strip reads retired
        if (t + 1 < T_) STAGE_STRIP(t + 1);                 // prefetch next strip

        for (int j = 0; j < 16; ++j) {       // GEMM1: 16x256 strip
            v8f acc = {};
            for (int kk = 0; kk < 4; ++kk) {
                v16h bf = *(const v16h*)(gW1F + (((kk * 16 + j) * 32) + lane) * 16);
                acc = __builtin_amdgcn_wmma_f32_16x16x32_f16(
                        false, afrag[kk], false, bf, (short)0, acc, false, false);
            }
            // store D tile directly in keys-GEMM A-fragment layout
            int kf = (j & 1) * 16 + p;                 // K within 32-wide ktile
            int e  = (kf & 7) | ((kf & 16) >> 1);      // element slot
            int lhi = (kf & 8) << 1;                   // lane' high-half bit
            for (int r = 0; r < 8; ++r)
                sOutFrag[j >> 1][(r + hi8) + lhi][e] = (_Float16)acc[r];
        }
        __syncthreads();
        for (int j2 = 0; j2 < 4; ++j2) {     // keys GEMM: 16x64
            v8f acc = {};
            for (int kk = 0; kk < 8; ++kk) {
                v16h af = *(const v16h*)(&sOutFrag[kk][lane][0]);   // one 32B LDS read
                v16h bf = *(const v16h*)(gW2F + (((kk * 4 + j2) * 32) + lane) * 16);
                acc = __builtin_amdgcn_wmma_f32_16x16x32_f16(
                        false, af, false, bf, (short)0, acc, false, false);
            }
            int col = j2 * 16 + p;
            float qv = gQS[col];
            int head = col >> 2;
            float lo = gLoff[(b * T_ + t) * NH_ + head];
            for (int r = 0; r < 8; ++r) {    // reduce DK=4 across 4 adjacent lanes
                float part = acc[r] * qv;
                part += __shfl_xor(part, 1, 32);
                part += __shfl_xor(part, 2, 32);
                if ((lane & 3) == 0)
                    sAttn[t][r + hi8][head] = part + lo;
            }
        }
        __syncthreads();
    }

    // ----- softmax over t per (pixel, head) -----
    for (int i = lane; i < 256; i += 32) {
        int pp = i >> 4, h = i & 15;
        float mx = -1e30f;
        for (int t = 0; t < T_; ++t) mx = fmaxf(mx, sAttn[t][pp][h]);
        float s = 0.f;
        for (int t = 0; t < T_; ++t) {
            float e = __expf(sAttn[t][pp][h] - mx);
            sAttn[t][pp][h] = e; s += e;
        }
        float inv = 1.f / s;
        for (int t = 0; t < T_; ++t) sAttn[t][pp][h] *= inv;
    }
    __syncthreads();

    // ----- pass 2: recompute GEMM1, accumulate attention-weighted values -----
    STAGE_STRIP(0);
    for (int t = 0; t < T_; ++t) {
        asm volatile("s_wait_asynccnt 0x0" ::: "memory");
        __syncthreads();

        v16h afrag[4];
        BUILD_AFRAG(afrag);
        asm volatile("s_wait_dscnt 0x0" ::: "memory");
        if (t + 1 < T_) STAGE_STRIP(t + 1);

        for (int j = 0; j < 16; ++j) {       // head = j (DM/NH == 16)
            v8f acc = {};
            for (int kk = 0; kk < 4; ++kk) {
                v16h bf = *(const v16h*)(gW1F + (((kk * 16 + j) * 32) + lane) * 16);
                acc = __builtin_amdgcn_wmma_f32_16x16x32_f16(
                        false, afrag[kk], false, bf, (short)0, acc, false, false);
            }
            int dm = j * 16 + p;
            float vb = gVbias[(b * T_ + t) * DM_ + dm];
            for (int r = 0; r < 8; ++r)
                sOacc[r + hi8][dm] += sAttn[t][r + hi8][j] * (acc[r] + vb);
        }
        __syncthreads();
    }

    // ----- MLP: [16 x 256] @ [256 x 128] -----
    v16h a3[8];
    for (int kk = 0; kk < 8; ++kk) {
        v16h f;
        for (int e = 0; e < 16; ++e)
            f[e] = (_Float16)sOacc[p][kk * 32 + frag_k(lane, e)];
        a3[kk] = f;
    }
    for (int j = 0; j < 8; ++j) {
        v8f acc = {};
        for (int kk = 0; kk < 8; ++kk) {
            v16h bf = *(const v16h*)(gW3F + (((kk * 8 + j) * 32) + lane) * 16);
            acc = __builtin_amdgcn_wmma_f32_16x16x32_f16(
                    false, a3[kk], false, bf, (short)0, acc, false, false);
        }
        int dm = j * 16 + p;
        float bm = b_mlp[dm];
        for (int r = 0; r < 8; ++r)
            gY[(size_t)(n0 + r + hi8) * DO_ + dm] = acc[r] + bm;
    }
    #undef STAGE_STRIP
    #undef BUILD_AFRAG
}

// ---------------------------------------------------------------------------
// deterministic BatchNorm reduction: per-block partial sums, then combine
// ---------------------------------------------------------------------------
__global__ void bn_reduce_kernel(const float* __restrict__ gY,
                                 float* __restrict__ gPart) {
    __shared__ float sS[256], sQ[256];
    int c = threadIdx.x & 127, half = threadIdx.x >> 7;
    int start = blockIdx.x * 128;                 // grid.x = NPIX/128 = 72
    float s = 0.f, s2 = 0.f;
    for (int r = half; r < 128; r += 2) {
        float v = gY[(size_t)(start + r) * DO_ + c];
        s += v; s2 += v * v;
    }
    sS[threadIdx.x] = s; sQ[threadIdx.x] = s2;
    __syncthreads();
    if (threadIdx.x < 128) {
        gPart[blockIdx.x * 256 + c]       = sS[c] + sS[c + 128];
        gPart[blockIdx.x * 256 + 128 + c] = sQ[c] + sQ[c + 128];
    }
}

__global__ void bn_final_kernel(const float* __restrict__ gPart,
                                const float* __restrict__ g_bn,
                                const float* __restrict__ b_bn,
                                float* __restrict__ gBnA,
                                float* __restrict__ gBnB) {
    int c = threadIdx.x;   // 128 threads
    float s = 0.f, s2 = 0.f;
    for (int i = 0; i < 72; ++i) { s += gPart[i * 256 + c]; s2 += gPart[i * 256 + 128 + c]; }
    float m = s / (float)NPIX;
    float var = s2 / (float)NPIX - m * m;
    float a = g_bn[c] * rsqrtf(var + EPS_);
    gBnA[c] = a;
    gBnB[c] = b_bn[c] - m * a;
}

// ---------------------------------------------------------------------------
// finalize: BN + ReLU + output GroupNorm (8-channel groups per lane) + store
// ---------------------------------------------------------------------------
__global__ void finalize_kernel(const float* __restrict__ gY,
                                const float* __restrict__ gBnA,
                                const float* __restrict__ gBnB,
                                const float* __restrict__ g_out,
                                const float* __restrict__ b_out,
                                float* __restrict__ out) {
    int tid = blockIdx.x * 256 + threadIdx.x;     // NPIX * 8 threads
    int n = tid >> 3, q = tid & 7;                // 16 channels per thread
    int b = n / HW_, hw = n - b * HW_;
    float v[16];
    for (int k = 0; k < 16; ++k) {
        int c = q * 16 + k;
        float y = gY[(size_t)n * DO_ + c] * gBnA[c] + gBnB[c];
        v[k] = fmaxf(y, 0.f);
    }
    for (int g2 = 0; g2 < 2; ++g2) {
        float s = 0.f, s2 = 0.f;
        for (int k = 0; k < 8; ++k) { float u = v[g2 * 8 + k]; s += u; s2 += u * u; }
        float m = s * 0.125f, var = s2 * 0.125f - m * m;
        float rs = rsqrtf(var + EPS_);
        for (int k = 0; k < 8; ++k) {
            int c = q * 16 + g2 * 8 + k;
            out[((size_t)b * DO_ + c) * HW_ + hw] =
                (v[g2 * 8 + k] - m) * rs * g_out[c] + b_out[c];
        }
    }
}

// ---------------------------------------------------------------------------
extern "C" void kernel_launch(void* const* d_in, const int* in_sizes, int n_in,
                              void* d_out, int out_size, void* d_ws, size_t ws_size,
                              hipStream_t stream) {
    const float* x        = (const float*)d_in[0];
    const int*   pos      = (const int*)  d_in[1];
    const float* g_in     = (const float*)d_in[2];
    const float* b_in     = (const float*)d_in[3];
    const float* W_inconv = (const float*)d_in[4];
    const float* b_inconv = (const float*)d_in[5];
    const float* Q        = (const float*)d_in[6];
    const float* W_k      = (const float*)d_in[7];
    const float* b_k      = (const float*)d_in[8];
    const float* W_mlp    = (const float*)d_in[9];
    const float* b_mlp    = (const float*)d_in[10];
    const float* g_bn     = (const float*)d_in[11];
    const float* b_bn     = (const float*)d_in[12];
    const float* g_out    = (const float*)d_in[13];
    const float* b_out    = (const float*)d_in[14];
    float* out = (float*)d_out;

    // workspace carve-up (all offsets 256B-aligned)
    char* ws = (char*)d_ws;
    size_t off = 0;
    float* MEAN  = (float*)(ws + off); off += (size_t)NPIX * NH_ * 4;        // 576 KB
    float* RSTD  = (float*)(ws + off); off += (size_t)NPIX * NH_ * 4;        // 576 KB
    float* VBIAS = (float*)(ws + off); off += (size_t)B_ * T_ * DM_ * 4;     //  96 KB
    float* LOFF  = (float*)(ws + off); off += (size_t)B_ * T_ * NH_ * 4;     //   6 KB
    float* QS    = (float*)(ws + off); off += 64 * 4;                        // 256 B
    _Float16* W1F = (_Float16*)(ws + off); off += (size_t)4 * 16 * 32 * 16 * 2;  // 64 KB
    _Float16* W2F = (_Float16*)(ws + off); off += (size_t)8 * 4  * 32 * 16 * 2;  // 32 KB
    _Float16* W3F = (_Float16*)(ws + off); off += (size_t)8 * 8  * 32 * 16 * 2;  // 64 KB
    float* Y     = (float*)(ws + off); off += (size_t)NPIX * DO_ * 4;        // 4.6 MB
    float* PART  = (float*)(ws + off); off += (size_t)72 * 256 * 4;          //  72 KB
    float* BNA   = (float*)(ws + off); off += 128 * 4;
    float* BNB   = (float*)(ws + off); off += 128 * 4;

    // prep: 64 + 24576 + 32768 + 16384 + 32768 + 1536 = 108096 threads used (140864 launched)
    prep_kernel<<<(140864 + 255) / 256, 256, 0, stream>>>(
        pos, b_inconv, Q, W_inconv, W_k, b_k, W_mlp, QS, VBIAS, LOFF, W1F, W2F, W3F);

    stats_kernel<<<dim3(NPIX / 256, NH_), 256, 0, stream>>>(x, MEAN, RSTD);

    main_kernel<<<NPIX / 16, 32, 0, stream>>>(
        x, g_in, b_in, b_mlp, MEAN, RSTD, VBIAS, LOFF, QS, W1F, W2F, W3F, Y);

    bn_reduce_kernel<<<NPIX / 128, 256, 0, stream>>>(Y, PART);
    bn_final_kernel<<<1, 128, 0, stream>>>(PART, g_bn, b_bn, BNA, BNB);

    finalize_kernel<<<(NPIX * 8) / 256, 256, 0, stream>>>(Y, BNA, BNB, g_out, b_out, out);
}
